// TransformerBlock_29721173689039
// MI455X (gfx1250) — matrix operand order repaired and checked by reading, not compiled, not run on hardware
//
#include <hip/hip_runtime.h>
#include <hip/hip_bf16.h>
#include <math.h>

typedef __attribute__((ext_vector_type(16))) _Float16 v16h;
typedef __attribute__((ext_vector_type(8)))  float    v8f;

#define B_   2
#define S_   2048
#define E_   768
#define H_   12
#define HD_  64
#define E4_  3072
#define M_   (B_*S_)   // 4096 rows

// ---------------------------------------------------------------- convert
__global__ void f32_to_f16_kernel(const float* __restrict__ src,
                                  _Float16* __restrict__ dst, int n) {
  int i = blockIdx.x * blockDim.x + threadIdx.x;
  int stride = gridDim.x * blockDim.x;
  for (; i < n; i += stride) dst[i] = (_Float16)src[i];
}

// ---------------------------------------------------------------- layernorm
// One 256-thread block per row of length E_=768 (3 elems/thread), f16 out.
__global__ __launch_bounds__(256)
void layernorm_f16_kernel(const float* __restrict__ x,
                          const float* __restrict__ scale,
                          const float* __restrict__ shift,
                          _Float16* __restrict__ out) {
  const int row = blockIdx.x;
  const float* xr = x + (size_t)row * E_;
  const int t = threadIdx.x;
  float v0 = xr[t], v1 = xr[t + 256], v2 = xr[t + 512];
  float s  = v0 + v1 + v2;
  float s2 = v0*v0 + v1*v1 + v2*v2;
  #pragma unroll
  for (int m = 16; m >= 1; m >>= 1) {
    s  += __shfl_xor(s,  m, 32);
    s2 += __shfl_xor(s2, m, 32);
  }
  __shared__ float red[2][8];
  int wave = t >> 5;
  if ((t & 31) == 0) { red[0][wave] = s; red[1][wave] = s2; }
  __syncthreads();
  if (wave == 0) {
    float a = red[0][t & 7];
    float b = red[1][t & 7];
    #pragma unroll
    for (int m = 4; m >= 1; m >>= 1) {
      a += __shfl_xor(a, m, 32);
      b += __shfl_xor(b, m, 32);
    }
    if (t == 0) { red[0][0] = a; red[1][0] = b; }
  }
  __syncthreads();
  float mean = red[0][0] * (1.0f / E_);
  float var  = red[1][0] * (1.0f / E_) - mean * mean;
  float rstd = rsqrtf(var + 1e-5f);
  _Float16* orow = out + (size_t)row * E_;
  orow[t]       = (_Float16)(scale[t]       * (v0 - mean) * rstd + shift[t]);
  orow[t + 256] = (_Float16)(scale[t + 256] * (v1 - mean) * rstd + shift[t + 256]);
  orow[t + 512] = (_Float16)(scale[t + 512] * (v2 - mean) * rstd + shift[t + 512]);
}

// ---------------------------------------------------------------- gemm
__device__ __forceinline__ float gelu_tanh_f(float x) {
  const float c = 0.7978845608028654f; // sqrt(2/pi)
  float x3 = x * x * x;
  return 0.5f * x * (1.0f + tanhf(c * (x + 0.044715f * x3)));
}

enum { MODE_QKV = 0, MODE_RESID = 1, MODE_GELU = 2 };

#define BM  128
#define BN  64
#define BK  32
#define LDA 40   // 32 + 8 pad (f16)
#define LDB 72   // 64 + 8 pad (f16)

// C[M,N] = A[M,K](f16,row-major) * Bm[K,N](f16,row-major).
// Block tile 128x64, 8 waves in 4x2 grid; each wave computes a 32x32 sub-tile
// (2 A-frags x 2 B-frags -> 4 WMMAs per K-step). Double-buffered LDS with
// register prefetch: one __syncthreads per K-step, next tile's global loads
// in flight while WMMAs execute.
template<int MODE>
__global__ __launch_bounds__(256)
void gemm_wmma_kernel(const _Float16* __restrict__ A,
                      const _Float16* __restrict__ Bm,
                      const float* __restrict__ bias,
                      const float* __restrict__ resid,
                      _Float16* __restrict__ outH,
                      float* __restrict__ outF,
                      int M, int N, int K) {
  __shared__ __align__(16) _Float16 As[2][BM * LDA];
  __shared__ __align__(16) _Float16 Bs[2][BK * LDB];
  const int tid  = threadIdx.x;
  const int lane = tid & 31;
  const int wave = tid >> 5;
  const int wm = wave >> 1;     // 0..3 -> 32-row band
  const int wn = wave & 1;      // 0..1 -> 32-col half
  const int row0 = blockIdx.y * BM;
  const int col0 = blockIdx.x * BN;
  const int r = lane & 15, g = lane >> 4;

  // staging maps: A tile 128x32 (512 uint4, 2/thread), B tile 32x64 (1/thread)
  const int a0r = tid >> 2;        const int a0c = (tid & 3) * 8;
  const int a1r = a0r + 64;        const int a1c = a0c;
  const int brr = tid >> 3;        const int bcc = (tid & 7) * 8;

  const _Float16* Aptr = A + (size_t)row0 * K;
  const _Float16* Bptr = Bm + col0;

  uint4 ra0 = *(const uint4*)(Aptr + (size_t)a0r * K + a0c);
  uint4 ra1 = *(const uint4*)(Aptr + (size_t)a1r * K + a1c);
  uint4 rb  = *(const uint4*)(Bptr + (size_t)brr * N + bcc);
  *(uint4*)&As[0][a0r * LDA + a0c] = ra0;
  *(uint4*)&As[0][a1r * LDA + a1c] = ra1;
  *(uint4*)&Bs[0][brr * LDB + bcc] = rb;
  __syncthreads();

  v8f acc00 = {}, acc01 = {}, acc10 = {}, acc11 = {};

  const int nsteps = K / BK;
  for (int s = 0; s < nsteps; ++s) {
    const int buf = s & 1;
    if (s + 1 < nsteps) {           // prefetch next K-tile into registers
      const int k0 = (s + 1) * BK;
      ra0 = *(const uint4*)(Aptr + (size_t)a0r * K + k0 + a0c);
      ra1 = *(const uint4*)(Aptr + (size_t)a1r * K + k0 + a1c);
      rb  = *(const uint4*)(Bptr + (size_t)(k0 + brr) * N + bcc);
    }

    v16h a0, a1;
    #pragma unroll
    for (int j = 0; j < 8; ++j) {
      int kk = 16 * g + 2 * j;
      const _Float16* p0 = &As[buf][(wm * 32 + r) * LDA + kk];
      const _Float16* p1 = &As[buf][(wm * 32 + 16 + r) * LDA + kk];
      a0[2 * j] = p0[0]; a0[2 * j + 1] = p0[1];
      a1[2 * j] = p1[0]; a1[2 * j + 1] = p1[1];
    }
    v16h b0, b1;
    #pragma unroll
    for (int h = 0; h < 16; ++h) {
      int kk = 16 * g + h;
      b0[h] = Bs[buf][kk * LDB + wn * 32 + r];
      b1[h] = Bs[buf][kk * LDB + wn * 32 + 16 + r];
    }
    acc00 = __builtin_amdgcn_wmma_f32_16x16x32_f16(false, a0, false, b0, (short)0, acc00, false, false);
    acc01 = __builtin_amdgcn_wmma_f32_16x16x32_f16(false, a0, false, b1, (short)0, acc01, false, false);
    acc10 = __builtin_amdgcn_wmma_f32_16x16x32_f16(false, a1, false, b0, (short)0, acc10, false, false);
    acc11 = __builtin_amdgcn_wmma_f32_16x16x32_f16(false, a1, false, b1, (short)0, acc11, false, false);

    if (s + 1 < nsteps) {           // stage into the other buffer
      const int nb = buf ^ 1;
      *(uint4*)&As[nb][a0r * LDA + a0c] = ra0;
      *(uint4*)&As[nb][a1r * LDA + a1c] = ra1;
      *(uint4*)&Bs[nb][brr * LDB + bcc] = rb;
    }
    __syncthreads();
  }

  #pragma unroll
  for (int am = 0; am < 2; ++am) {
    #pragma unroll
    for (int bn = 0; bn < 2; ++bn) {
      const v8f* accp = (am == 0) ? (bn == 0 ? &acc00 : &acc01)
                                  : (bn == 0 ? &acc10 : &acc11);
      #pragma unroll
      for (int i = 0; i < 8; ++i) {
        int m = row0 + wm * 32 + am * 16 + i + 8 * g;  // C/D: VGPR i holds row i+8g
        int n = col0 + wn * 32 + bn * 16 + r;
        float c = (*accp)[i];
        if (MODE == MODE_QKV) {
          int b = m >> 11, ss = m & (S_ - 1);
          int hh = n >> 6, d = n & 63;
          outH[((size_t)(b * H_ + hh) * S_ + ss) * HD_ + d] = (_Float16)c;
        } else if (MODE == MODE_RESID) {
          size_t o = (size_t)m * N + n;
          outF[o] = resid[o] + c + bias[n];
        } else { // MODE_GELU
          outH[(size_t)m * N + n] = (_Float16)gelu_tanh_f(c + bias[n]);
        }
      }
    }
  }
}

// ---------------------------------------------------------------- flash attention
// Q,K,V: [B,H,S,HD] f16. Each wave owns a 16-row Q tile, streams 32-key tiles.
// Scores 16x32 via 4 WMMAs, online softmax in C-layout (half-wave shfl reductions),
// P relaid to A-fragment via per-wave LDS, P*V via 4 WMMAs. ctx -> [B,S,E] f16.
__global__ __launch_bounds__(128)
void flash_attn_kernel(const _Float16* __restrict__ Q,
                       const _Float16* __restrict__ Kmat,
                       const _Float16* __restrict__ V,
                       _Float16* __restrict__ ctx) {
  __shared__ _Float16 Ps[4][16 * 32];
  const int lane = threadIdx.x & 31;
  const int wave = threadIdx.x >> 5;
  const int r = lane & 15, g = lane >> 4;
  const int bh = blockIdx.y;
  const int q0 = (blockIdx.x * 4 + wave) * 16;
  const _Float16* Qp = Q    + (size_t)bh * S_ * HD_;
  const _Float16* Kp = Kmat + (size_t)bh * S_ * HD_;
  const _Float16* Vp = V    + (size_t)bh * S_ * HD_;

  // Q fragments (k=0..31 and 32..63), fold in softmax scale 1/sqrt(64)=0.125
  v16h aq0, aq1;
  #pragma unroll
  for (int j = 0; j < 8; ++j) {
    int kk = 16 * g + 2 * j;
    aq0[2 * j]     = (_Float16)((float)Qp[(size_t)(q0 + r) * HD_ + kk]      * 0.125f);
    aq0[2 * j + 1] = (_Float16)((float)Qp[(size_t)(q0 + r) * HD_ + kk + 1]  * 0.125f);
    aq1[2 * j]     = (_Float16)((float)Qp[(size_t)(q0 + r) * HD_ + kk + 32] * 0.125f);
    aq1[2 * j + 1] = (_Float16)((float)Qp[(size_t)(q0 + r) * HD_ + kk + 33] * 0.125f);
  }

  v8f acc0 = {}, acc1 = {}, acc2 = {}, acc3 = {};
  float mrow[8], lrow[8];
  #pragma unroll
  for (int i = 0; i < 8; ++i) { mrow[i] = -__builtin_inff(); lrow[i] = 0.0f; }

  const int kend = q0 + 15;
  for (int kb = 0; kb <= kend; kb += 32) {
    // ---- scores: S(16x32) = Qs(16x64) @ K_tile^T(64x32)
    v8f s0 = {}, s1 = {};
    v16h bk;
    #pragma unroll
    for (int h = 0; h < 16; ++h) bk[h] = Kp[(size_t)(kb + r) * HD_ + 16 * g + h];
    s0 = __builtin_amdgcn_wmma_f32_16x16x32_f16(false, aq0, false, bk, (short)0, s0, false, false);
    #pragma unroll
    for (int h = 0; h < 16; ++h) bk[h] = Kp[(size_t)(kb + r) * HD_ + 32 + 16 * g + h];
    s0 = __builtin_amdgcn_wmma_f32_16x16x32_f16(false, aq1, false, bk, (short)0, s0, false, false);
    #pragma unroll
    for (int h = 0; h < 16; ++h) bk[h] = Kp[(size_t)(kb + 16 + r) * HD_ + 16 * g + h];
    s1 = __builtin_amdgcn_wmma_f32_16x16x32_f16(false, aq0, false, bk, (short)0, s1, false, false);
    #pragma unroll
    for (int h = 0; h < 16; ++h) bk[h] = Kp[(size_t)(kb + 16 + r) * HD_ + 32 + 16 * g + h];
    s1 = __builtin_amdgcn_wmma_f32_16x16x32_f16(false, aq1, false, bk, (short)0, s1, false, false);

    // ---- causal mask + online softmax (row i+8g lives in lane half g)
    #pragma unroll
    for (int i = 0; i < 8; ++i) {
      int qrow = q0 + i + 8 * g;
      float e0 = ((kb + r)      <= qrow) ? s0[i] : -__builtin_inff();
      float e1 = ((kb + 16 + r) <= qrow) ? s1[i] : -__builtin_inff();
      float rm = fmaxf(e0, e1);
      #pragma unroll
      for (int msk = 8; msk >= 1; msk >>= 1) rm = fmaxf(rm, __shfl_xor(rm, msk, 32));
      float mnew = fmaxf(mrow[i], rm);
      float p0 = __expf(e0 - mnew);
      float p1 = __expf(e1 - mnew);
      float corr = __expf(mrow[i] - mnew);
      float rs = p0 + p1;
      #pragma unroll
      for (int msk = 8; msk >= 1; msk >>= 1) rs += __shfl_xor(rs, msk, 32);
      lrow[i] = lrow[i] * corr + rs;
      mrow[i] = mnew;
      acc0[i] *= corr; acc1[i] *= corr; acc2[i] *= corr; acc3[i] *= corr;
      Ps[wave][(i + 8 * g) * 32 + r]      = (_Float16)p0;
      Ps[wave][(i + 8 * g) * 32 + 16 + r] = (_Float16)p1;
    }
    // wave-local LDS ordering (no block barrier: waves have different trip counts)
    asm volatile("s_wait_dscnt 0" ::: "memory");

    // ---- P (16x32) as A fragment
    v16h aP;
    #pragma unroll
    for (int j = 0; j < 8; ++j) {
      int kk = 16 * g + 2 * j;
      aP[2 * j]     = Ps[wave][r * 32 + kk];
      aP[2 * j + 1] = Ps[wave][r * 32 + kk + 1];
    }
    // ---- ctx += P @ V_tile(32x64), 4 column chunks of 16
    v16h bv;
    #pragma unroll
    for (int h = 0; h < 16; ++h) bv[h] = Vp[(size_t)(kb + 16 * g + h) * HD_ + r];
    acc0 = __builtin_amdgcn_wmma_f32_16x16x32_f16(false, aP, false, bv, (short)0, acc0, false, false);
    #pragma unroll
    for (int h = 0; h < 16; ++h) bv[h] = Vp[(size_t)(kb + 16 * g + h) * HD_ + 16 + r];
    acc1 = __builtin_amdgcn_wmma_f32_16x16x32_f16(false, aP, false, bv, (short)0, acc1, false, false);
    #pragma unroll
    for (int h = 0; h < 16; ++h) bv[h] = Vp[(size_t)(kb + 16 * g + h) * HD_ + 32 + r];
    acc2 = __builtin_amdgcn_wmma_f32_16x16x32_f16(false, aP, false, bv, (short)0, acc2, false, false);
    #pragma unroll
    for (int h = 0; h < 16; ++h) bv[h] = Vp[(size_t)(kb + 16 * g + h) * HD_ + 48 + r];
    acc3 = __builtin_amdgcn_wmma_f32_16x16x32_f16(false, aP, false, bv, (short)0, acc3, false, false);
  }

  // ---- normalize + store ctx back to [B,S,E] (merge heads)
  const int b = bh / H_, h = bh % H_;
  #pragma unroll
  for (int i = 0; i < 8; ++i) {
    int qrow = q0 + i + 8 * g;
    float rinv = 1.0f / lrow[i];
    size_t base = ((size_t)(b * S_ + qrow)) * E_ + h * HD_;
    ctx[base + r]      = (_Float16)(acc0[i] * rinv);
    ctx[base + 16 + r] = (_Float16)(acc1[i] * rinv);
    ctx[base + 32 + r] = (_Float16)(acc2[i] * rinv);
    ctx[base + 48 + r] = (_Float16)(acc3[i] * rinv);
  }
}

// ---------------------------------------------------------------- launch
extern "C" void kernel_launch(void* const* d_in, const int* in_sizes, int n_in,
                              void* d_out, int out_size, void* d_ws, size_t ws_size,
                              hipStream_t stream) {
  const float* x    = (const float*)d_in[0];
  const float* Wq   = (const float*)d_in[1];
  const float* Wk   = (const float*)d_in[2];
  const float* Wv   = (const float*)d_in[3];
  const float* Wo   = (const float*)d_in[4];
  const float* bo   = (const float*)d_in[5];
  const float* W1   = (const float*)d_in[6];
  const float* b1   = (const float*)d_in[7];
  const float* W2   = (const float*)d_in[8];
  const float* b2   = (const float*)d_in[9];
  const float* ln1s = (const float*)d_in[10];
  const float* ln1b = (const float*)d_in[11];
  const float* ln2s = (const float*)d_in[12];
  const float* ln2b = (const float*)d_in[13];
  float* out = (float*)d_out;

  char* ws = (char*)d_ws;
  const size_t EE  = (size_t)E_ * E_;        // 589824
  const size_t EW  = (size_t)E_ * E4_;       // 2359296
  const size_t ME  = (size_t)M_ * E_;        // 3145728

  size_t off = 0;
  _Float16* Wq_h = (_Float16*)(ws + off); off += EE * 2;
  _Float16* Wk_h = (_Float16*)(ws + off); off += EE * 2;
  _Float16* Wv_h = (_Float16*)(ws + off); off += EE * 2;
  _Float16* Wo_h = (_Float16*)(ws + off); off += EE * 2;
  _Float16* W1_h = (_Float16*)(ws + off); off += EW * 2;
  _Float16* W2_h = (_Float16*)(ws + off); off += EW * 2;
  _Float16* hA   = (_Float16*)(ws + off); off += ME * 2;      // reused as h2 after attn
  size_t qkvRegion = off;
  _Float16* Qh   = (_Float16*)(ws + off); off += ME * 2;
  _Float16* Kh   = (_Float16*)(ws + off); off += ME * 2;
  _Float16* Vh   = (_Float16*)(ws + off); off += ME * 2;
  _Float16* ctxh = (_Float16*)(ws + off); off += ME * 2;
  float*    x1   = (float*)(ws + off);    off += ME * 4;
  _Float16* h2   = hA;                                        // reuse
  _Float16* ff1  = (_Float16*)(ws + qkvRegion);               // reuse Q/K/V/ctx (4*ME f16)
  (void)ws_size; (void)n_in; (void)in_sizes; (void)out_size;

  // 1) weights -> f16
  f32_to_f16_kernel<<<(EE + 255) / 256, 256, 0, stream>>>(Wq, Wq_h, (int)EE);
  f32_to_f16_kernel<<<(EE + 255) / 256, 256, 0, stream>>>(Wk, Wk_h, (int)EE);
  f32_to_f16_kernel<<<(EE + 255) / 256, 256, 0, stream>>>(Wv, Wv_h, (int)EE);
  f32_to_f16_kernel<<<(EE + 255) / 256, 256, 0, stream>>>(Wo, Wo_h, (int)EE);
  f32_to_f16_kernel<<<(EW + 255) / 256, 256, 0, stream>>>(W1, W1_h, (int)EW);
  f32_to_f16_kernel<<<(EW + 255) / 256, 256, 0, stream>>>(W2, W2_h, (int)EW);

  // 2) LN1
  layernorm_f16_kernel<<<M_, 256, 0, stream>>>(x, ln1s, ln1b, hA);

  // 3) QKV projections (fused head-layout epilogue)
  dim3 gEE(E_ / BN, M_ / BM);       // (12, 32)
  gemm_wmma_kernel<MODE_QKV><<<gEE, 256, 0, stream>>>(hA, Wq_h, nullptr, nullptr, Qh, nullptr, M_, E_, E_);
  gemm_wmma_kernel<MODE_QKV><<<gEE, 256, 0, stream>>>(hA, Wk_h, nullptr, nullptr, Kh, nullptr, M_, E_, E_);
  gemm_wmma_kernel<MODE_QKV><<<gEE, 256, 0, stream>>>(hA, Wv_h, nullptr, nullptr, Vh, nullptr, M_, E_, E_);

  // 4) causal flash attention
  flash_attn_kernel<<<dim3(S_ / 64, B_ * H_), 128, 0, stream>>>(Qh, Kh, Vh, ctxh);

  // 5) output projection + bias + residual -> x1 (f32)
  gemm_wmma_kernel<MODE_RESID><<<gEE, 256, 0, stream>>>(ctxh, Wo_h, bo, x, nullptr, x1, M_, E_, E_);

  // 6) LN2
  layernorm_f16_kernel<<<M_, 256, 0, stream>>>(x1, ln2s, ln2b, h2);

  // 7) MLP up + GELU (f16 out)
  gemm_wmma_kernel<MODE_GELU><<<dim3(E4_ / BN, M_ / BM), 256, 0, stream>>>(h2, W1_h, b1, nullptr, ff1, nullptr, M_, E4_, E_);

  // 8) MLP down + bias + residual -> final f32 output
  gemm_wmma_kernel<MODE_RESID><<<gEE, 256, 0, stream>>>(ff1, W2_h, b2, x1, nullptr, out, M_, E_, E4_);
}
